// MobiusGRU_16595753632518
// MI455X (gfx1250) — compile-verified, matrix operand order checked by reading, not run on hardware
//
#include <hip/hip_runtime.h>
#include <hip/hip_bf16.h>

// ---- problem constants (from the reference) ----
#define TT 256
#define BB 64
#define HH 512
#define LL 2
#define G3 (3*HH)
#define EPSF 1e-15f

// LDS pitches (padded to avoid ds bank conflicts on b128 row loads)
#define PF 516   // float pitch for the 64xH f32 work tile
#define PB 520   // bf16 pitch for the 64xH bf16 matrices (520*2B = 65*16B)

typedef __attribute__((ext_vector_type(16))) __bf16 v16bf;
typedef __attribute__((ext_vector_type(8)))  float  v8f;
typedef __attribute__((ext_vector_type(4)))  unsigned int u32x4;

union BFr { v16bf v; u32x4 q[2]; };

#define RED16(x) do { x += __shfl_xor(x, 8, 16); x += __shfl_xor(x, 4, 16); \
                      x += __shfl_xor(x, 2, 16); x += __shfl_xor(x, 1, 16); } while (0)
#define RED8(x)  do { x += __shfl_xor(x, 4, 8);  x += __shfl_xor(x, 2, 8);  \
                      x += __shfl_xor(x, 1, 8); } while (0)

__device__ __forceinline__ float artanh_(float x) {
  x = fminf(fmaxf(x, -1.f + 1e-7f), 1.f - 1e-7f);
  return 0.5f * (log1pf(x) - log1pf(-x));
}
__device__ __forceinline__ float sigmoid_(float x) { return 1.f / (1.f + expf(-x)); }
// Mobius add element given per-row scalars x2=|x|^2, y2=|y|^2, xy=<x,y> (C=1)
__device__ __forceinline__ float madd_el(float x, float y, float x2, float y2, float xy) {
  float num = (1.f + 2.f * xy + y2) * x + (1.f - x2) * y;
  float den = 1.f + 2.f * xy + x2 * y2;
  return num / fmaxf(den, EPSF);
}

// One 16-wide column tile (nt) of  M(64xHH) = A(64xHH,bf16) * Wg(HH rows x HH,bf16)^T
// using v_wmma_f32_16x16x32_bf16.  A in LDS (pitch PB), Wg row-major in global,
// output f32 tile in LDS (pitch PF).
__device__ __forceinline__ void gemm_tilecol(const __bf16* __restrict__ Abf,
                                             const __bf16* __restrict__ Wg,
                                             float* __restrict__ M,
                                             int nt, int lane) {
  const int l16 = lane & 15;
  const int kh  = lane >> 4;
  const __bf16* brow = Wg + (size_t)(nt * 16 + l16) * HH;
#pragma unroll
  for (int mt = 0; mt < 4; ++mt) {
    const __bf16* arow = Abf + (mt * 16 + l16) * PB;
    v8f acc = {0.f, 0.f, 0.f, 0.f, 0.f, 0.f, 0.f, 0.f};
#pragma unroll
    for (int kb = 0; kb < 16; ++kb) {
      const int k0 = kb * 32 + kh * 8;
      BFr a, b;
      a.q[0] = *(const u32x4*)(arow + k0);        // K = k0..k0+7    (elems 0..7)
      a.q[1] = *(const u32x4*)(arow + k0 + 16);   // K = k0+16..+23  (elems 8..15)
      b.q[0] = *(const u32x4*)(brow + k0);
      b.q[1] = *(const u32x4*)(brow + k0 + 16);
      acc = __builtin_amdgcn_wmma_f32_16x16x32_bf16(false, a.v, false, b.v,
                                                    (short)0, acc, false, false);
    }
    float* orow = M + (mt * 16 + kh * 8) * PF + nt * 16 + l16;
#pragma unroll
    for (int v = 0; v < 8; ++v) orow[v * PF] = acc[v];   // C layout: m = v + 8*kh
  }
}

// ------------------------------------------------------------------
// Kernel 0: fp32 -> bf16 weight conversion (Wih and Whh, all layers)
// ------------------------------------------------------------------
__global__ void cvt_kernel(const float* __restrict__ a, const float* __restrict__ b,
                           __bf16* __restrict__ ab, __bf16* __restrict__ bb, int n) {
  int i = blockIdx.x * blockDim.x + threadIdx.x;
  int stride = gridDim.x * blockDim.x;
  for (; i < n; i += stride) { ab[i] = (__bf16)a[i]; bb[i] = (__bf16)b[i]; }
}

// ------------------------------------------------------------------
// Kernel 1 (GPU-wide): precompute p_{t,g} = mobius_matvec(W_ig, x_t)
// for all timesteps.  grid = (T, 3 gates), 512 threads (16 waves).
// ------------------------------------------------------------------
__global__ __launch_bounds__(512) void proj_kernel(const float* __restrict__ X,   // [T][B][H]
                                                   const __bf16* __restrict__ Wih, // [3H][H]
                                                   float* __restrict__ gx) {       // [T][3][B][H]
  extern __shared__ char smem[];
  float*  m   = (float*)smem;                             // 64*PF f32
  __bf16* xbf = (__bf16*)(smem + 64 * PF * 4);            // 64*PB bf16
  float*  xn2 = (float*)(smem + 64 * PF * 4 + 64 * PB * 2);
  float*  mn2 = xn2 + 64;

  const int t = blockIdx.x, g = blockIdx.y;
  const int tid = threadIdx.x, row = tid >> 3, sub = tid & 7;

  // load x_t, convert to bf16, accumulate |x|^2 per row
  const float* xrow = X + ((size_t)t * BB + row) * HH;
  float s = 0.f;
  for (int j = 0; j < 64; ++j) {
    int c = sub + (j << 3);
    float v = xrow[c];
    xbf[row * PB + c] = (__bf16)v;
    s += v * v;
  }
  RED8(s);
  if (!sub) xn2[row] = s;
  __syncthreads();

  // mx = x W_g^T  (M=64, N=512, K=512): 16 waves x 2 column tiles
  {
    const int lane = tid & 31, wave = tid >> 5;
    const __bf16* Wg = Wih + (size_t)g * HH * HH;
    gemm_tilecol(xbf, Wg, m, wave, lane);
    gemm_tilecol(xbf, Wg, m, wave + 16, lane);
  }
  __syncthreads();

  // |mx|^2 per row
  float s2 = 0.f;
  for (int j = 0; j < 64; ++j) { int c = sub + (j << 3); float v = m[row * PF + c]; s2 += v * v; }
  RED8(s2);
  if (!sub) mn2[row] = s2;
  __syncthreads();

  // p = tanh(mn/xn * artanh(xn)) * mx / mn
  float mn = sqrtf(mn2[row] + EPSF), xn = sqrtf(xn2[row] + EPSF);
  float sc = tanhf(mn / xn * artanh_(xn)) / mn;
  float* orow = gx + (((size_t)t * 3 + g) * BB + row) * HH;
  for (int j = 0; j < 64; ++j) { int c = sub + (j << 3); orow[c] = sc * m[row * PF + c]; }
}

// ------------------------------------------------------------------
// Recurrent helpers (1024 threads, 16 lanes per row, 64 rows)
// ------------------------------------------------------------------
// Computes m := mobius_add(mobius_add(mobius_matvec_scaled(m_raw), p), b)
// where m_raw = A @ Wg^T is computed here.  Leaves in t0 = |result|^2 and
// t1 = <result, -h> for the caller.
__device__ void mobius_transform(const __bf16* __restrict__ Abf, const float* __restrict__ xn2,
                                 const __bf16* __restrict__ Wg, const float* __restrict__ p,
                                 const float* __restrict__ bg,
                                 float* __restrict__ m, const __bf16* __restrict__ hbf,
                                 float* t0, float* t1, float* t2,
                                 float* u0, float* u1, float* u2) {
  const int tid = threadIdx.x, row = tid >> 4, sub = tid & 15;
  const int lane = tid & 31, wave = tid >> 5;

  gemm_tilecol(Abf, Wg, m, wave, lane);     // 32 waves -> 32 column tiles
  __syncthreads();

  { // |m|^2
    float s = 0.f;
    for (int j = 0; j < 32; ++j) { int c = sub + (j << 4); float v = m[row * PF + c]; s += v * v; }
    RED16(s);
    if (!sub) t0[row] = s;
  }
  __syncthreads();
  { // mv = s*m (mobius_matvec scale); accumulate <mv,p>, |p|^2
    float mn = sqrtf(t0[row] + EPSF), xn = sqrtf(xn2[row] + EPSF);
    float sc = tanhf(mn / xn * artanh_(xn)) / mn;
    float a1 = 0.f, a2 = 0.f;
    for (int j = 0; j < 32; ++j) {
      int c = sub + (j << 4);
      float mv = sc * m[row * PF + c]; m[row * PF + c] = mv;
      float pv = p[row * HH + c];
      a1 += mv * pv; a2 += pv * pv;
    }
    RED16(a1); RED16(a2);
    if (!sub) { t1[row] = a1; t2[row] = a2; }
  }
  __syncthreads();
  { // a = mobius_add(mv, p); accumulate |a|^2, <a,b>, |b|^2
    float mn = sqrtf(t0[row] + EPSF), xn = sqrtf(xn2[row] + EPSF);
    float sc = tanhf(mn / xn * artanh_(xn)) / mn;
    float x2 = sc * sc * t0[row], xy = t1[row], y2 = t2[row];
    float a0 = 0.f, a1 = 0.f, a2 = 0.f;
    for (int j = 0; j < 32; ++j) {
      int c = sub + (j << 4);
      float a = madd_el(m[row * PF + c], p[row * HH + c], x2, y2, xy);
      m[row * PF + c] = a;
      float bv = bg[c];
      a0 += a * a; a1 += a * bv; a2 += bv * bv;
    }
    RED16(a0); RED16(a1); RED16(a2);
    if (!sub) { u0[row] = a0; u1[row] = a1; u2[row] = a2; }
  }
  __syncthreads();
  { // v = mobius_add(a, b); accumulate |v|^2 and <v,-h>
    float x2 = u0[row], xy = u1[row], y2 = u2[row];
    float a0 = 0.f, a1 = 0.f;
    for (int j = 0; j < 32; ++j) {
      int c = sub + (j << 4);
      float v = madd_el(m[row * PF + c], bg[c], x2, y2, xy);
      m[row * PF + c] = v;
      a0 += v * v;
      a1 -= v * (float)hbf[row * PB + c];
    }
    RED16(a0); RED16(a1);
    if (!sub) { t0[row] = a0; t1[row] = a1; }
  }
  __syncthreads();
}

// ------------------------------------------------------------------
// Kernel 2: persistent single-workgroup recurrence over T steps.
// ------------------------------------------------------------------
__global__ __launch_bounds__(1024) void recur_kernel(const float* __restrict__ gx,   // [T][3][B][H]
                                                     const __bf16* __restrict__ Whh, // [3H][H]
                                                     const float* __restrict__ bias, // [3][H]
                                                     float* __restrict__ out_seq,    // [T][B][H]
                                                     float* __restrict__ hlast,      // [B][H]
                                                     float* __restrict__ zbuf) {     // [B][H]
  extern __shared__ char smem[];
  float*  m    = (float*)smem;                                      // 64*PF f32
  __bf16* hbf  = (__bf16*)(smem + 64 * PF * 4);                     // 64*PB
  __bf16* rhbf = (__bf16*)(smem + 64 * PF * 4 + 64 * PB * 2);       // 64*PB
  float*  scr  = (float*)(smem + 64 * PF * 4 + 2 * 64 * PB * 2);
  float* hn2 = scr;        float* rhn2 = scr + 64;
  float* t0  = scr + 128;  float* t1   = scr + 192;  float* t2 = scr + 256;
  float* u0  = scr + 320;  float* u1   = scr + 384;  float* u2 = scr + 448;

  const int tid = threadIdx.x, row = tid >> 4, sub = tid & 15;

  // h0 = 0
  for (int j = 0; j < 32; ++j) { int c = sub + (j << 4); hbf[row * PB + c] = (__bf16)0.f; }
  if (tid < 64) hn2[tid] = 0.f;
  __syncthreads();

  for (int t = 0; t < TT; ++t) {
    const float* pr = gx + (((size_t)t * 3 + 0) * BB) * HH;
    const float* ph = gx + (((size_t)t * 3 + 1) * BB) * HH;
    const float* pz = gx + (((size_t)t * 3 + 2) * BB) * HH;

    // ---- z = sigmoid(logmap0(one_transform(W_hz, h, W_iz, x, b_z))) ----
    mobius_transform(hbf, hn2, Whh + (size_t)2 * HH * HH, pz, bias + 2 * HH,
                     m, hbf, t0, t1, t2, u0, u1, u2);
    {
      float yn = sqrtf(t0[row] + EPSF);
      float sc = artanh_(yn) / yn;
      for (int j = 0; j < 32; ++j) { int c = sub + (j << 4);
        zbuf[row * HH + c] = sigmoid_(sc * m[row * PF + c]); }
    }
    __syncthreads();

    // ---- r gate (kept in m) ----
    mobius_transform(hbf, hn2, Whh, pr, bias,
                     m, hbf, t0, t1, t2, u0, u1, u2);
    {
      float yn = sqrtf(t0[row] + EPSF);
      float sc = artanh_(yn) / yn;
      for (int j = 0; j < 32; ++j) { int c = sub + (j << 4);
        m[row * PF + c] = sigmoid_(sc * m[row * PF + c]); }
    }
    __syncthreads();

    // ---- rh = mobius_pw_mul(r, h) ----
    { // wx = r*h, |wx|^2
      float a0 = 0.f;
      for (int j = 0; j < 32; ++j) { int c = sub + (j << 4);
        float wx = m[row * PF + c] * (float)hbf[row * PB + c];
        m[row * PF + c] = wx; a0 += wx * wx; }
      RED16(a0);
      if (!sub) t0[row] = a0;
    }
    __syncthreads();
    { // rh = tanh(wn/hn*artanh(hn)) * wx / wn -> bf16
      float wn = sqrtf(t0[row] + EPSF), hn = sqrtf(hn2[row] + EPSF);
      float s2 = tanhf(wn / hn * artanh_(hn)) / wn;
      for (int j = 0; j < 32; ++j) { int c = sub + (j << 4);
        rhbf[row * PB + c] = (__bf16)(s2 * m[row * PF + c]); }
      if (!sub) rhn2[row] = s2 * s2 * t0[row];
    }
    __syncthreads();

    // ---- h_tilde = one_transform(W_hhid, rh, W_ihid, x, b_h) ----
    mobius_transform(rhbf, rhn2, Whh + (size_t)1 * HH * HH, ph, bias + HH,
                     m, hbf, t0, t1, t2, u0, u1, u2);
    // now: m = h_tilde, t0 = |ht|^2, t1 = <ht,-h>

    { // delta = mobius_add(-h, ht); |delta|^2
      float x2 = hn2[row], y2 = t0[row], xy = t1[row];
      float a0 = 0.f;
      for (int j = 0; j < 32; ++j) { int c = sub + (j << 4);
        float d = madd_el(-(float)hbf[row * PB + c], m[row * PF + c], x2, y2, xy);
        m[row * PF + c] = d; a0 += d * d; }
      RED16(a0);
      if (!sub) u0[row] = a0;
    }
    __syncthreads();
    { // wx = z * delta; |wx|^2
      float a0 = 0.f;
      for (int j = 0; j < 32; ++j) { int c = sub + (j << 4);
        float wx = zbuf[row * HH + c] * m[row * PF + c];
        m[row * PF + c] = wx; a0 += wx * wx; }
      RED16(a0);
      if (!sub) u1[row] = a0;
    }
    __syncthreads();
    { // zd = pw_mul scale; accumulate <h, zd>
      float dn = sqrtf(u0[row] + EPSF), wn = sqrtf(u1[row] + EPSF);
      float s3 = tanhf(wn / dn * artanh_(dn)) / wn;
      float a0 = 0.f;
      for (int j = 0; j < 32; ++j) { int c = sub + (j << 4);
        float zd = s3 * m[row * PF + c];
        m[row * PF + c] = zd;
        a0 += (float)hbf[row * PB + c] * zd; }
      RED16(a0);
      if (!sub) u2[row] = a0;
    }
    __syncthreads();
    { // h_new = mobius_add(h, zd); emit out[t]; refresh hn2
      float dn = sqrtf(u0[row] + EPSF), wn = sqrtf(u1[row] + EPSF);
      float s3 = tanhf(wn / dn * artanh_(dn)) / wn;
      float x2 = hn2[row], y2 = s3 * s3 * u1[row], xy = u2[row];
      float a0 = 0.f;
      float* orow = out_seq + ((size_t)t * BB + row) * HH;
      for (int j = 0; j < 32; ++j) { int c = sub + (j << 4);
        float hv = (float)hbf[row * PB + c];
        float hn_new = madd_el(hv, m[row * PF + c], x2, y2, xy);
        orow[c] = hn_new;
        hbf[row * PB + c] = (__bf16)hn_new;
        if (t == TT - 1) hlast[row * HH + c] = hn_new;
        a0 += hn_new * hn_new; }
      RED16(a0);
      if (!sub) hn2[row] = a0;
    }
    __syncthreads();
  }
}

// ------------------------------------------------------------------
extern "C" void kernel_launch(void* const* d_in, const int* in_sizes, int n_in,
                              void* d_out, int out_size, void* d_ws, size_t ws_size,
                              hipStream_t stream) {
  (void)in_sizes; (void)n_in; (void)out_size; (void)ws_size;
  const float* input = (const float*)d_in[0];   // [T][B][H]
  const float* Wih   = (const float*)d_in[1];   // [L][3H][H]
  const float* Whh   = (const float*)d_in[2];   // [L][3H][H]
  const float* bias  = (const float*)d_in[3];   // [L][3][H]
  float* out = (float*)d_out;                   // [T][B][H] ++ [L][B][H]

  char* ws = (char*)d_ws;
  const size_t nW = (size_t)LL * G3 * HH;       // 1.57M weights each
  __bf16* Wih_bf = (__bf16*)ws;  ws += nW * 2;
  __bf16* Whh_bf = (__bf16*)ws;  ws += nW * 2;
  float*  gx   = (float*)ws;     ws += (size_t)TT * 3 * BB * HH * 4;  // ~100 MB
  float*  out0 = (float*)ws;     ws += (size_t)TT * BB * HH * 4;      // ~33 MB
  float*  zbuf = (float*)ws;     ws += (size_t)BB * HH * 4;

  const size_t PROJ_SMEM  = (size_t)64 * PF * 4 + (size_t)64 * PB * 2 + 2 * 64 * 4;
  const size_t RECUR_SMEM = (size_t)64 * PF * 4 + 2 * (size_t)64 * PB * 2 + 8 * 64 * 4;

  cvt_kernel<<<512, 256, 0, stream>>>(Wih, Whh, Wih_bf, Whh_bf, (int)nW);

  for (int l = 0; l < LL; ++l) {
    const float* xs = (l == 0) ? input : out0;
    float* od = (l == LL - 1) ? out : out0;
    proj_kernel<<<dim3(TT, 3), 512, PROJ_SMEM, stream>>>(
        xs, Wih_bf + (size_t)l * G3 * HH, gx);
    recur_kernel<<<1, 1024, RECUR_SMEM, stream>>>(
        gx, Whh_bf + (size_t)l * G3 * HH, bias + (size_t)l * 3 * HH,
        od, out + (size_t)TT * BB * HH + (size_t)l * BB * HH, zbuf);
  }
}
